// ViTARCMHSA_7576322310859
// MI455X (gfx1250) — matrix operand order repaired and checked by reading, not compiled, not run on hardware
//
#include <hip/hip_runtime.h>

#define B_  8
#define S_  1024
#define E_  768
#define H_  12
#define D_  64
#define N3_ 2304
#define M_  (B_ * S_)   // 8192

typedef __attribute__((ext_vector_type(16))) __bf16 v16bf;
typedef __attribute__((ext_vector_type(8)))  float  v8f;
typedef int v4i __attribute__((vector_size(16)));

union BFPack { v16bf v; uint4 u[2]; };

// ---- gfx1250 async global->LDS copy (ASYNCcnt path), with sync fallback ----
#if __has_builtin(__builtin_amdgcn_global_load_async_to_lds_b128) && \
    __has_builtin(__builtin_amdgcn_s_wait_asynccnt)
#define HAVE_ASYNC 1
#else
#define HAVE_ASYNC 0
#endif

__device__ __forceinline__ void cp16_async(unsigned short* l, const unsigned short* g) {
#if HAVE_ASYNC
  __builtin_amdgcn_global_load_async_to_lds_b128(
      (__attribute__((address_space(1))) v4i*)(g),
      (__attribute__((address_space(3))) v4i*)(l), 0, 0);
#else
  *(uint4*)l = *(const uint4*)g;
#endif
}
__device__ __forceinline__ void wait_async_le2() {
#if HAVE_ASYNC
  __builtin_amdgcn_s_wait_asynccnt(2);
#endif
}
__device__ __forceinline__ void wait_async_le0() {
#if HAVE_ASYNC
  __builtin_amdgcn_s_wait_asynccnt(0);
#endif
}

__device__ __forceinline__ unsigned short f2bf(float f) {
  unsigned int u = __float_as_uint(f);
  u += 0x7FFFu + ((u >> 16) & 1u);   // round-to-nearest-even
  return (unsigned short)(u >> 16);
}

// 16 contiguous bf16 (32 bytes) -> one WMMA B-fragment lane payload
__device__ __forceinline__ v16bf ld_frag32(const unsigned short* p) {
  BFPack r;
  r.u[0] = *(const uint4*)(p);
  r.u[1] = *(const uint4*)(p + 8);
  return r.v;
}
// A-fragment lane payload: K = kbase..kbase+7 then K = kbase+16..kbase+23
__device__ __forceinline__ v16bf ld_frag16x2(const unsigned short* p0, const unsigned short* p1) {
  BFPack r;
  r.u[0] = *(const uint4*)(p0);
  r.u[1] = *(const uint4*)(p1);
  return r.v;
}

__device__ __forceinline__ v8f wmma_bf16(v16bf a, v16bf b, v8f c) {
  return __builtin_amdgcn_wmma_f32_16x16x32_bf16(false, a, false, b, (short)0, c, false, false);
}

// xor-shuffle within the wave via ds_swizzle (group-of-32: and=0x1f, or=0, xor=MASK)
template <int MASK>
__device__ __forceinline__ float swz(float v) {
  return __int_as_float(__builtin_amdgcn_ds_swizzle(__float_as_int(v), (MASK << 10) | 0x1f));
}
__device__ __forceinline__ float red_max16(float v) {  // reduce within 16-lane half
  v = fmaxf(v, swz<1>(v)); v = fmaxf(v, swz<2>(v));
  v = fmaxf(v, swz<4>(v)); v = fmaxf(v, swz<8>(v));
  return v;
}
__device__ __forceinline__ float red_add16(float v) {
  v += swz<1>(v); v += swz<2>(v); v += swz<4>(v); v += swz<8>(v);
  return v;
}

// ---------------------------------------------------------------------------
// Kernel 1: qkv = x @ w_qkv.T   (f32 in -> bf16 q/k/v out, v stored transposed)
// block 256 (8 waves), tile 128(M) x 64(N), BK=32  (staging converts, so sync)
// ---------------------------------------------------------------------------
__global__ __launch_bounds__(256) void qkv_gemm_kernel(
    const float* __restrict__ x, const float* __restrict__ wqkv,
    unsigned short* __restrict__ qbf, unsigned short* __restrict__ kbf,
    unsigned short* __restrict__ vbf) {
  __shared__ alignas(16) unsigned short As[128 * 32];
  __shared__ alignas(16) unsigned short Bs[64 * 32];
  const int tid = threadIdx.x;
  const int L = tid & 31, w = tid >> 5;
  const int g = L >> 4, ln = L & 15;
  const int m0 = blockIdx.y * 128, n0 = blockIdx.x * 64;

  v8f acc[4] = {};

  for (int kk = 0; kk < E_; kk += 32) {
#pragma unroll
    for (int i = 0; i < 4; ++i) {
      int lin = (tid + i * 256) * 4;
      int row = lin >> 5, col = lin & 31;
      float4 v = *(const float4*)(x + (size_t)(m0 + row) * E_ + kk + col);
      unsigned int lo = (unsigned)f2bf(v.x) | ((unsigned)f2bf(v.y) << 16);
      unsigned int hi = (unsigned)f2bf(v.z) | ((unsigned)f2bf(v.w) << 16);
      *(uint2*)(&As[row * 32 + col]) = make_uint2(lo, hi);
    }
#pragma unroll
    for (int i = 0; i < 2; ++i) {
      int lin = (tid + i * 256) * 4;
      int row = lin >> 5, col = lin & 31;
      float4 v = *(const float4*)(wqkv + (size_t)(n0 + row) * E_ + kk + col);
      unsigned int lo = (unsigned)f2bf(v.x) | ((unsigned)f2bf(v.y) << 16);
      unsigned int hi = (unsigned)f2bf(v.z) | ((unsigned)f2bf(v.w) << 16);
      *(uint2*)(&Bs[row * 32 + col]) = make_uint2(lo, hi);
    }
    __syncthreads();
    const unsigned short* ap = As + (w * 16 + ln) * 32 + g * 16;
    v16bf a = ld_frag32(ap);
#pragma unroll
    for (int nt = 0; nt < 4; ++nt) {
      v16bf b = ld_frag32(Bs + (nt * 16 + ln) * 32 + g * 16);
      acc[nt] = wmma_bf16(a, b, acc[nt]);
    }
    __syncthreads();
  }

#pragma unroll
  for (int nt = 0; nt < 4; ++nt) {
    int n = n0 + nt * 16 + ln;
    int which = n / E_;
    int rem = n % E_;
    int hh = rem >> 6, d = rem & 63;
#pragma unroll
    for (int r = 0; r < 8; ++r) {
      int m = m0 + w * 16 + r + 8 * g;
      int b = m >> 10, s = m & 1023;
      unsigned short val = f2bf(acc[nt][r]);
      size_t bh = (size_t)(b * H_ + hh);
      if (which == 0)       qbf[(bh * S_ + s) * D_ + d] = val;
      else if (which == 1)  kbf[(bh * S_ + s) * D_ + d] = val;
      else                  vbf[(bh * D_ + d) * S_ + s] = val;   // V transposed (D,S)
    }
  }
}

// ---------------------------------------------------------------------------
// Kernel 2: flash attention, two-slope ALiBi bias.
// Block = 256 thr (8 waves) on one (b,h): 128 q rows; K/V 32-key chunk staged
// in LDS once per block via double-buffered async copies (8x less K/V traffic).
// ---------------------------------------------------------------------------
__global__ __launch_bounds__(256) void attn_kernel(
    const unsigned short* __restrict__ qbf, const unsigned short* __restrict__ kbf,
    const unsigned short* __restrict__ vbf, const float* __restrict__ dist,
    unsigned short* __restrict__ attnbf) {
  __shared__ alignas(16) unsigned short Ks[2][32 * 64];   // [key][d]
  __shared__ alignas(16) unsigned short Vs[2][64 * 32];   // [d][key]
  __shared__ alignas(16) unsigned short Plds[8][16 * 32]; // per-wave P buffer

  const int tid = threadIdx.x;
  const int L = tid & 31, w = tid >> 5;
  const int g = L >> 4, ln = L & 15;
  const int qbase = blockIdx.x * 128 + w * 16;
  const int bh = blockIdx.y;            // b*H + h
  const int h = bh % H_;
  const int b = bh / H_;

  // slopes: get_slopes(12, se) -> exponent e(h) = h (h<8) else 2*(h-8)
  const float eh = (h < 8) ? (float)h : (float)(2 * (h - 8));
  const float sl = -exp2f(-1.0f - eh);
  const float sr = -exp2f(-0.5f - eh);

  const unsigned short* qptr = qbf + (size_t)bh * S_ * D_;
  const unsigned short* kptr = kbf + (size_t)bh * S_ * D_;
  const unsigned short* vptr = vbf + (size_t)bh * D_ * S_;

  // Q A-fragments: resident for the whole loop
  v16bf qa[2];
#pragma unroll
  for (int f = 0; f < 2; ++f) {
    const unsigned short* p = qptr + (size_t)(qbase + ln) * D_ + f * 32 + g * 16;
    qa[f] = ld_frag32(p);
  }

  v8f o[4] = {};
  float mrow[8], lrow[8];
#pragma unroll
  for (int r = 0; r < 8; ++r) { mrow[r] = -3.0e38f; lrow[r] = 0.0f; }

  // per-thread 16B slice of the K and V chunk tiles
  const int lin  = tid * 8;
  const int krow = lin >> 6, kcol = lin & 63;   // K tile: 32x64
  const int vrow = lin >> 5, vcol = lin & 31;   // V tile: 64x32

  // prologue: stage chunk 0 into buffer 0
  cp16_async(&Ks[0][krow * 64 + kcol], kptr + (size_t)krow * D_ + kcol);
  cp16_async(&Vs[0][vrow * 32 + vcol], vptr + (size_t)vrow * S_ + vcol);

  for (int t = 0; t < S_ / 32; ++t) {
    const int kc = t * 32;
    const int buf = t & 1;
    if (t + 1 < S_ / 32) {   // prefetch next chunk into the other buffer
      const int kn = kc + 32, bn = buf ^ 1;
      cp16_async(&Ks[bn][krow * 64 + kcol], kptr + (size_t)(kn + krow) * D_ + kcol);
      cp16_async(&Vs[bn][vrow * 32 + vcol], vptr + (size_t)vrow * S_ + kn + vcol);
      wait_async_le2();      // chunk t complete; chunk t+1 still in flight
    } else {
      wait_async_le0();
    }
    __syncthreads();

    // ---- scores: 16q x 32k from LDS K tile ----
    const unsigned short* kt = &Ks[buf][0];
    v8f s0 = {}, s1 = {};
    s0 = wmma_bf16(qa[0], ld_frag32(kt + ln * 64 + g * 16), s0);
    s0 = wmma_bf16(qa[1], ld_frag32(kt + ln * 64 + 32 + g * 16), s0);
    s1 = wmma_bf16(qa[0], ld_frag32(kt + (16 + ln) * 64 + g * 16), s1);
    s1 = wmma_bf16(qa[1], ld_frag32(kt + (16 + ln) * 64 + 32 + g * 16), s1);

    // ---- bias: triu(sr*rel)+tril(sl*rel); diagonal gets (sr+sl)*rel ----
    const int j0 = kc + ln, j1 = j0 + 16;
#pragma unroll
    for (int r = 0; r < 8; ++r) {
      int i = qbase + r + 8 * g;
      float rel0 = dist[i * S_ + j0];
      float rel1 = dist[i * S_ + j1];
      float w0 = (j0 > i) ? sr : ((j0 == i) ? (sr + sl) : sl);
      float w1 = (j1 > i) ? sr : ((j1 == i) ? (sr + sl) : sl);
      s0[r] += rel0 * w0;
      s1[r] += rel1 * w1;
    }
    // ---- online softmax ----
    float p0[8], p1[8];
#pragma unroll
    for (int r = 0; r < 8; ++r) {
      float mx = red_max16(fmaxf(s0[r], s1[r]));
      float mnew = fmaxf(mrow[r], mx);
      float alpha = __expf(mrow[r] - mnew);
      mrow[r] = mnew;
      p0[r] = __expf(s0[r] - mnew);
      p1[r] = __expf(s1[r] - mnew);
      float rs = red_add16(p0[r] + p1[r]);
      lrow[r] = lrow[r] * alpha + rs;
#pragma unroll
      for (int f = 0; f < 4; ++f) o[f][r] *= alpha;
    }
    // ---- C-layout P -> A-layout via per-wave LDS buffer ----
    unsigned short* pw = &Plds[w][0];
#pragma unroll
    for (int r = 0; r < 8; ++r) {
      int rl = r + 8 * g;
      pw[rl * 32 + ln]      = f2bf(p0[r]);
      pw[rl * 32 + 16 + ln] = f2bf(p1[r]);
    }
    __syncthreads();
    v16bf pa = ld_frag32(pw + ln * 32 + g * 16);
    // ---- O += P @ V from LDS V tile ----
#pragma unroll
    for (int f = 0; f < 4; ++f) {
      o[f] = wmma_bf16(pa, ld_frag32(&Vs[buf][(f * 16 + ln) * 32 + g * 16]), o[f]);
    }
    __syncthreads();   // reads done before next chunk's copies land in this buffer
  }

  // normalize + store as (B,S,H*D) bf16
  unsigned short* op = attnbf + (size_t)b * S_ * E_ + h * D_;
#pragma unroll
  for (int f = 0; f < 4; ++f) {
#pragma unroll
    for (int r = 0; r < 8; ++r) {
      int row = qbase + r + 8 * g;
      op[(size_t)row * E_ + f * 16 + ln] = f2bf(o[f][r] / lrow[r]);
    }
  }
}

// ---------------------------------------------------------------------------
// Kernel 3: out = attn @ w_proj.T + b_proj
// A (bf16) staged via async copies overlapping the f32->bf16 B conversion.
// ---------------------------------------------------------------------------
__global__ __launch_bounds__(256) void proj_gemm_kernel(
    const unsigned short* __restrict__ attn, const float* __restrict__ wproj,
    const float* __restrict__ bproj, float* __restrict__ out) {
  __shared__ alignas(16) unsigned short As[128 * 32];
  __shared__ alignas(16) unsigned short Bs[64 * 32];
  const int tid = threadIdx.x;
  const int L = tid & 31, w = tid >> 5;
  const int g = L >> 4, ln = L & 15;
  const int m0 = blockIdx.y * 128, n0 = blockIdx.x * 64;

  v8f acc[4] = {};

  for (int kk = 0; kk < E_; kk += 32) {
    // A tile: pure copy -> async to LDS (overlaps with B conversion below)
#pragma unroll
    for (int i = 0; i < 2; ++i) {
      int lin = (tid + i * 256) * 8;
      int row = lin >> 5, col = lin & 31;
      cp16_async(&As[row * 32 + col], attn + (size_t)(m0 + row) * E_ + kk + col);
    }
    // B tile: f32 -> bf16 conversion (VALU work hides async latency)
#pragma unroll
    for (int i = 0; i < 2; ++i) {
      int lin = (tid + i * 256) * 4;
      int row = lin >> 5, col = lin & 31;
      float4 v = *(const float4*)(wproj + (size_t)(n0 + row) * E_ + kk + col);
      unsigned int lo = (unsigned)f2bf(v.x) | ((unsigned)f2bf(v.y) << 16);
      unsigned int hi = (unsigned)f2bf(v.z) | ((unsigned)f2bf(v.w) << 16);
      *(uint2*)(&Bs[row * 32 + col]) = make_uint2(lo, hi);
    }
    wait_async_le0();
    __syncthreads();
    const unsigned short* ap = As + (w * 16 + ln) * 32 + g * 16;
    v16bf a = ld_frag32(ap);
#pragma unroll
    for (int nt = 0; nt < 4; ++nt) {
      v16bf b = ld_frag32(Bs + (nt * 16 + ln) * 32 + g * 16);
      acc[nt] = wmma_bf16(a, b, acc[nt]);
    }
    __syncthreads();
  }

#pragma unroll
  for (int nt = 0; nt < 4; ++nt) {
    int n = n0 + nt * 16 + ln;
    float bias = bproj[n];
#pragma unroll
    for (int r = 0; r < 8; ++r) {
      int m = m0 + w * 16 + r + 8 * g;
      out[(size_t)m * E_ + n] = acc[nt][r] + bias;
    }
  }
}

// ---------------------------------------------------------------------------
extern "C" void kernel_launch(void* const* d_in, const int* in_sizes, int n_in,
                              void* d_out, int out_size, void* d_ws, size_t ws_size,
                              hipStream_t stream) {
  (void)in_sizes; (void)n_in; (void)out_size; (void)ws_size;
  const float* x     = (const float*)d_in[0];
  const float* dist  = (const float*)d_in[1];
  const float* wqkv  = (const float*)d_in[2];
  const float* wproj = (const float*)d_in[3];
  const float* bproj = (const float*)d_in[4];
  float* out = (float*)d_out;

  const size_t elems = (size_t)B_ * H_ * S_ * D_;   // 6291456 per tensor
  unsigned short* qbf  = (unsigned short*)d_ws;
  unsigned short* kbf  = qbf + elems;
  unsigned short* vbf  = kbf + elems;               // transposed (B,H,D,S)
  unsigned short* attn = vbf + elems;               // (B,S,E) bf16

  qkv_gemm_kernel<<<dim3(N3_ / 64, M_ / 128), 256, 0, stream>>>(x, wqkv, qbf, kbf, vbf);
  attn_kernel<<<dim3(S_ / 128, B_ * H_), 256, 0, stream>>>(qbf, kbf, vbf, dist, attn);
  proj_gemm_kernel<<<dim3(E_ / 64, M_ / 128), 256, 0, stream>>>(attn, wproj, bproj, out);
}
